// GNNSage_dev_5446018532030
// MI455X (gfx1250) — matrix operand order repaired and checked by reading, not compiled, not run on hardware
//
#include <hip/hip_runtime.h>
#include <hip/hip_bf16.h>
#include <math.h>

typedef __attribute__((ext_vector_type(16))) _Float16 v16h;
typedef __attribute__((ext_vector_type(2)))  _Float16 v2h;
typedef __attribute__((ext_vector_type(8)))  float    v8f;

#define D 128
#define LDS_STRIDE 136   // 128 + 8 f16 pad -> 272B rows, conflict-free column reads

// ---------------- utility kernels ----------------

__global__ void k_zero_f32(float* __restrict__ p, long n) {
  long i = (long)blockIdx.x * blockDim.x + threadIdx.x;
  long stride = (long)gridDim.x * blockDim.x;
  for (; i < n; i += stride) p[i] = 0.0f;
}

__global__ void k_degree(const int* __restrict__ dst, float* __restrict__ deg, int E) {
  int e = blockIdx.x * blockDim.x + threadIdx.x;
  if (e < E) unsafeAtomicAdd(&deg[dst[e]], 1.0f);
}

// one wave per edge; 32 lanes x float4 = 128 features. All traffic L2-resident.
__global__ void k_agg(const float* __restrict__ X, const int* __restrict__ src,
                      const int* __restrict__ dst, float* __restrict__ agg, int E) {
  int tid  = blockIdx.x * blockDim.x + threadIdx.x;
  int wid  = tid >> 5;
  int lane = tid & 31;
  if (wid >= E) return;
  int s = src[wid], d = dst[wid];
  float4 v = ((const float4*)(X + (long)s * D))[lane];
  float* ad = agg + (long)d * D + lane * 4;
  unsafeAtomicAdd(ad + 0, v.x);
  unsafeAtomicAdd(ad + 1, v.y);
  unsafeAtomicAdd(ad + 2, v.z);
  unsafeAtomicAdd(ad + 3, v.w);
}

// Pack a 128x128 f32 weight matrix (row-major [K][N]) into fragment-major f16:
// index = (((s*8 + t)*32 + lane)*16 + elem), so each lane's B fragment for
// (k-step s, n-tile t) is one contiguous 32B chunk. elem 2j/2j+1 map to
// K = 32s + 16*(j>>2) + 8*(lane>>4) + 2*(j&3)(+1), N = 16t + (lane&15)  (ISA 16-bit B layout).
__global__ void k_convert_w(const float* __restrict__ W, _Float16* __restrict__ Wp) {
  int o = blockIdx.x * blockDim.x + threadIdx.x;
  if (o >= D * D) return;
  int e = o & 15, lane = (o >> 4) & 31, t = (o >> 9) & 7, s = (o >> 12) & 3;
  int j = e >> 1, odd = e & 1;
  int K  = 32 * s + ((j & 4) << 2) + 8 * (lane >> 4) + 2 * (j & 3) + odd;
  int Nc = 16 * t + (lane & 15);
  Wp[o] = (_Float16)W[K * D + Nc];
}

// ---------------- layer 1: WMMA GEMM ----------------
// Block = 256 threads = 8 waves; block owns a 128-row node tile.
// Wave w computes rows [16w,16w+16) x all 128 cols = 8 accumulators (v8f each).
// Pass 0: X @ W_self1 ; Pass 1: (agg/deg) @ W_neigh1 accumulated into same C.
__launch_bounds__(256)
__global__ void k_layer1(const float* __restrict__ X, const float* __restrict__ agg,
                         const float* __restrict__ deg,
                         const _Float16* __restrict__ Wsp, const _Float16* __restrict__ Wnp,
                         const float* __restrict__ b1, float* __restrict__ H, int N) {
  __shared__ _Float16 Atile[128 * LDS_STRIDE];   // 34,816 B

  const int tid  = threadIdx.x;
  const int wave = tid >> 5;
  const int lane = tid & 31;
  const int m0   = blockIdx.x * 128;

  v8f acc[8] = {};

  for (int pass = 0; pass < 2; ++pass) {
    __syncthreads();                       // previous pass done reading LDS
    const float* Ag = pass ? agg : X;
    // stage 128x128 f32 -> f16 LDS (mean-scale applied on neighbor pass)
    for (int it = 0; it < 16; ++it) {
      int i4  = it * 256 + tid;            // 4096 float4 elements in tile
      int row = i4 >> 5;
      int c4  = i4 & 31;
      int m   = m0 + row;
      float4 v = make_float4(0.f, 0.f, 0.f, 0.f);
      float scale = 1.0f;
      if (m < N) {
        v = ((const float4*)(Ag + (long)m * D))[c4];
        if (pass) scale = 1.0f / fmaxf(deg[m], 1.0f);
      }
      _Float16* dp = &Atile[row * LDS_STRIDE + c4 * 4];
      v2h p0, p1;
      p0[0] = (_Float16)(v.x * scale); p0[1] = (_Float16)(v.y * scale);
      p1[0] = (_Float16)(v.z * scale); p1[1] = (_Float16)(v.w * scale);
      *(v2h*)(dp)     = p0;
      *(v2h*)(dp + 2) = p1;
    }
    __syncthreads();

    const _Float16* Wp = pass ? Wnp : Wsp;
    const int r = wave * 16 + (lane & 15);            // A row this lane serves
    const int kbaseA = (lane >> 4) * 8;               // ISA 16-bit A layout
    const _Float16* arow = &Atile[r * LDS_STRIDE];

    #pragma unroll
    for (int s = 0; s < 4; ++s) {
      const int k0 = s * 32;
      v16h a;
      #pragma unroll
      for (int j = 0; j < 8; ++j) {
        // pairs at K offsets {kb, kb+2, kb+4, kb+6, 16+kb, ..., 16+kb+6}
        int k = k0 + ((j & 4) << 2) + kbaseA + 2 * (j & 3);
        v2h p = *(const v2h*)(arow + k);
        a[2 * j]     = p[0];
        a[2 * j + 1] = p[1];
      }
      #pragma unroll
      for (int t = 0; t < 8; ++t) {
        v16h b = *(const v16h*)(Wp + (((s * 8 + t) * 32 + lane) << 4));
        acc[t] = __builtin_amdgcn_wmma_f32_16x16x32_f16(
            false, a, false, b, (short)0, acc[t], false, false);
      }
    }
  }

  // epilogue: bias + relu, store f32 (C layout: VGPR v -> M = v + 8*(lane>>4))
  const int hi   = lane >> 4;
  const int col0 = lane & 15;
  #pragma unroll
  for (int t = 0; t < 8; ++t) {
    int col = t * 16 + col0;
    float bias = b1[col];
    #pragma unroll
    for (int v = 0; v < 8; ++v) {
      int m = m0 + wave * 16 + hi * 8 + v;
      if (m < N) H[(long)m * D + col] = fmaxf(acc[t][v] + bias, 0.0f);
    }
  }
}

// ---------------- layer 2: GEMV + sigmoid (wave32 per node) ----------------
__global__ void k_layer2(const float* __restrict__ H, const float* __restrict__ agg,
                         const float* __restrict__ deg, const float* __restrict__ Ws2,
                         const float* __restrict__ Wn2, const float* __restrict__ b2,
                         float* __restrict__ out, int N) {
  int tid  = blockIdx.x * blockDim.x + threadIdx.x;
  int node = tid >> 5;
  int lane = tid & 31;
  if (node >= N) return;
  float scale = 1.0f / fmaxf(deg[node], 1.0f);
  float4 hv = ((const float4*)(H   + (long)node * D))[lane];
  float4 av = ((const float4*)(agg + (long)node * D))[lane];
  float4 ws = ((const float4*)Ws2)[lane];
  float4 wn = ((const float4*)Wn2)[lane];
  float acc = hv.x * ws.x + hv.y * ws.y + hv.z * ws.z + hv.w * ws.w
            + scale * (av.x * wn.x + av.y * wn.y + av.z * wn.z + av.w * wn.w);
  #pragma unroll
  for (int off = 16; off >= 1; off >>= 1)
    acc += __shfl_xor(acc, off, 32);
  if (lane == 0) out[node] = 1.0f / (1.0f + __expf(-(acc + b2[0])));
}

// ---------------- host launch ----------------

extern "C" void kernel_launch(void* const* d_in, const int* in_sizes, int n_in,
                              void* d_out, int out_size, void* d_ws, size_t ws_size,
                              hipStream_t stream) {
  (void)n_in; (void)out_size; (void)ws_size;
  const float* X   = (const float*)d_in[0];
  const int*   src = (const int*)d_in[1];
  const int*   dst = (const int*)d_in[2];
  const float* Ws1 = (const float*)d_in[3];
  const float* Wn1 = (const float*)d_in[4];
  const float* b1  = (const float*)d_in[5];
  const float* Ws2 = (const float*)d_in[6];
  const float* Wn2 = (const float*)d_in[7];
  const float* b2  = (const float*)d_in[8];
  float* out = (float*)d_out;

  const int N = in_sizes[0] / D;
  const int E = in_sizes[1];

  // workspace carve-up (256B aligned; ~10.4 MB total)
  char* ws = (char*)d_ws;
  size_t off = 0;
  auto carve = [&](size_t bytes) {
    char* p = ws + off;
    off = (off + bytes + 255) & ~(size_t)255;
    return p;
  };
  float*    deg = (float*)carve((size_t)N * sizeof(float));
  float*    agg = (float*)carve((size_t)N * D * sizeof(float));
  float*    H   = (float*)carve((size_t)N * D * sizeof(float));
  _Float16* Wsp = (_Float16*)carve((size_t)D * D * sizeof(_Float16));
  _Float16* Wnp = (_Float16*)carve((size_t)D * D * sizeof(_Float16));

  const long nAgg = (long)N * D;
  int zbDeg = (int)((N + 255) / 256);
  int zbAgg = (int)((nAgg + 255) / 256);

  k_zero_f32<<<zbDeg, 256, 0, stream>>>(deg, N);
  k_zero_f32<<<zbAgg, 256, 0, stream>>>(agg, nAgg);
  k_convert_w<<<(D * D + 255) / 256, 256, 0, stream>>>(Ws1, Wsp);
  k_convert_w<<<(D * D + 255) / 256, 256, 0, stream>>>(Wn1, Wnp);
  k_degree<<<(E + 255) / 256, 256, 0, stream>>>(dst, deg, E);

  // layer 1: aggregate X, then fused GEMM+bias+relu
  int aggBlocks = (int)(((long)E * 32 + 255) / 256);
  k_agg<<<aggBlocks, 256, 0, stream>>>(X, src, dst, agg, E);
  k_layer1<<<(N + 127) / 128, 256, 0, stream>>>(X, agg, deg, Wsp, Wnp, b1, H, N);

  // layer 2: re-aggregate h, then GEMV + sigmoid
  k_zero_f32<<<zbAgg, 256, 0, stream>>>(agg, nAgg);
  k_agg<<<aggBlocks, 256, 0, stream>>>(H, src, dst, agg, E);
  int l2Blocks = (int)(((long)N * 32 + 255) / 256);
  k_layer2<<<l2Blocks, 256, 0, stream>>>(H, agg, deg, Ws2, Wn2, b2, out, N);
}
